// CGCNNLayer_46102178955276
// MI455X (gfx1250) — matrix operand order repaired and checked by reading, not compiled, not run on hardware
//
#include <hip/hip_runtime.h>
#include <hip/hip_bf16.h>
#include <math.h>

// ---------------- problem constants ----------------
#define N_NODES   20000
#define N_EDGES   480000
#define HIDDEN    128
#define ZDIM      384            // 2*HIDDEN + EDGE_FEAT
#define NOUT      256            // 2*HIDDEN
#define BN_EPS    1e-5f

#define TILE_M    64             // edges per block
#define KSTEPS    (ZDIM / 32)    // 12 bf16-WMMA K steps
#define NBLOCKS   (N_EDGES / TILE_M)   // 7500 exactly

// TDM-padded B tile in LDS: row stride = 192 data dwords + 3 pads * 4 dwords = 204 dwords = 816 B
// 816 % 16 == 0 (aligned b128), 204 % 64 == 12 -> 16 lanes' b128 reads are bank-conflict-free.
#define B_ROW_BYTES   816
#define BS_BYTES      (NOUT * B_ROW_BYTES)   // 208896 dynamic LDS

typedef __attribute__((ext_vector_type(16))) __bf16 v16bf;
typedef __attribute__((ext_vector_type(8)))  float  v8f;
typedef __attribute__((ext_vector_type(4)))  unsigned int v4u;
typedef __attribute__((ext_vector_type(8)))  int    v8i;
typedef __attribute__((ext_vector_type(4)))  int    v4i;

// ---------------- workspace layout (bytes) ----------------
#define WS_WT     0            // Wt bf16 [256][384]  196608
#define WS_GSUM   196608       // f32[256]
#define WS_GSQ    197632       // f32[256]
#define WS_SCALE  198656       // f32[256]
#define WS_SHIFT  199680       // f32[256]
#define WS_AGG    200704       // f32[20000*128]   10240000
#define WS_Z      10440704     // bf16[480000*256] 245760000

// ================= kernel 1: prep ==========
__global__ void cgcnn_prep(const float* __restrict__ W, __bf16* __restrict__ Wt,
                           float* __restrict__ gsum, float* __restrict__ gsq,
                           float* __restrict__ agg) {
    const int i = blockIdx.x * 256 + threadIdx.x;   // grid covers 2,560,000
    if (i < NOUT * ZDIM) {
        const int n = i / ZDIM;
        const int k = i - n * ZDIM;
        Wt[i] = (__bf16)W[(size_t)k * NOUT + n];    // Wt[n][k] = W[k][n]
    }
    if (i < NOUT)            gsum[i] = 0.0f;
    else if (i < 2 * NOUT)   gsq[i - NOUT] = 0.0f;
    if (i < N_NODES * HIDDEN) agg[i] = 0.0f;
}

// ================= kernel 2: edge GEMM (TDM-staged B, bf16 WMMA) + BN stats ==========
__global__ __launch_bounds__(256)
void cgcnn_gemm(const float* __restrict__ h, const float* __restrict__ e,
                const int* __restrict__ src, const int* __restrict__ dst,
                const __bf16* __restrict__ Wt, __bf16* __restrict__ zb,
                float* __restrict__ gsum, float* __restrict__ gsq) {
    __shared__ __bf16 As[TILE_M][ZDIM + 8];   // row stride 784B (bank stagger for A)
    __shared__ float s_sum[NOUT];
    __shared__ float s_sq[NOUT];
    __shared__ int   s_src[TILE_M];
    __shared__ int   s_dst[TILE_M];
    extern __shared__ __align__(16) char dynsmem[];   // BS_BYTES: TDM-padded Wt tile

    const int tid = threadIdx.x;
    const int eb  = blockIdx.x * TILE_M;

    // ---- issue TDM: Wt[256][384] bf16 -> LDS with hardware padding (async, overlaps A-stage) ----
    if (tid == 0) {
        const unsigned lds_off = (unsigned)(uintptr_t)dynsmem;          // LDS byte offset
        const unsigned long long ga = (unsigned long long)(uintptr_t)Wt;
        v4u g0;
        g0[0] = 1u;                                   // count=1, is_restore=0, gather off
        g0[1] = lds_off;                              // lds_addr [63:32]
        g0[2] = (unsigned)ga;                         // global_addr [95:64]
        g0[3] = (unsigned)(ga >> 32) | (2u << 30);    // global_addr hi + type=2 [127:126]
        v8i g1;
        g1[0] = (int)((1u << 16)      // data_size = 2 bytes
                    | (1u << 20)      // pad_enable
                    | (5u << 22)      // pad_interval: 64 dwords
                    | (3u << 25));    // pad_amount: 4 dwords
        g1[1] = (int)(384u << 16);    // tensor_dim0[15:0] in [63:48]
        g1[2] = (int)(256u << 16);    // tensor_dim0 hi=0, tensor_dim1=256
        g1[3] = (int)(384u << 16);    // tensor_dim1 hi=0, tile_dim0=384
        g1[4] = (int)256;             // tile_dim1=256, tile_dim2=0
        g1[5] = (int)384;             // tensor_dim0_stride = 384
        g1[6] = 0;
        g1[7] = 0;
        v4i gz4 = {0, 0, 0, 0};
        v8i gz8 = {0, 0, 0, 0, 0, 0, 0, 0};
        __builtin_amdgcn_tensor_load_to_lds(g0, g1, gz4, gz4, gz8, 0);
    }

    s_sum[tid] = 0.0f;
    s_sq[tid]  = 0.0f;
    if (tid < TILE_M) {
        s_src[tid] = src[eb + tid];
        s_dst[tid] = dst[eb + tid];
    }
    __syncthreads();   // publish s_src/s_dst

    // ---- stage A tile: [h_src | h_dst | e] -> bf16 LDS (overlapped with TDM) ----
    for (int i = tid; i < TILE_M * (ZDIM / 4); i += 256) {
        const int row = i / (ZDIM / 4);
        const int q   = i - row * (ZDIM / 4);
        const int col = q * 4;
        float4 v;
        if (col < HIDDEN) {
            v = *(const float4*)(h + (size_t)s_src[row] * HIDDEN + col);
        } else if (col < 2 * HIDDEN) {
            v = *(const float4*)(h + (size_t)s_dst[row] * HIDDEN + (col - HIDDEN));
        } else {
            v = *(const float4*)(e + (size_t)(eb + row) * HIDDEN + (col - 2 * HIDDEN));
        }
        alignas(8) __bf16 p[4];
        p[0] = (__bf16)v.x; p[1] = (__bf16)v.y; p[2] = (__bf16)v.z; p[3] = (__bf16)v.w;
        *(uint2*)(&As[row][col]) = *(const uint2*)p;
    }

    __builtin_amdgcn_s_wait_tensorcnt(0);   // no-op for waves that issued nothing
    __syncthreads();

    // ---- wave tiling: 8 waves as 2(M) x 4(N); wave does 32x64 = 2x4 WMMA tiles ----
    const int lane   = tid & 31;
    const int l16    = lane & 15;
    const int g      = lane >> 4;        // half-wave group
    const int wave   = tid >> 5;
    const int wave_m = wave & 1;
    const int wave_n = wave >> 1;

    // per-lane B row base (bytes into padded LDS tile)
    const char* Bbase = (const char*)dynsmem;
    int brow[4];
#pragma unroll
    for (int nt = 0; nt < 4; ++nt) {
        const int n = wave_n * 64 + nt * 16 + l16;
        brow[nt] = B_ROW_BYTES * n + 32 * g;
    }

    v8f acc[2][4];
#pragma unroll
    for (int mt = 0; mt < 2; ++mt)
#pragma unroll
        for (int nt = 0; nt < 4; ++nt) {
            v8f z = {0.f,0.f,0.f,0.f,0.f,0.f,0.f,0.f};
            acc[mt][nt] = z;
        }

#pragma unroll 1
    for (int kk = 0; kk < KSTEPS; ++kk) {
        const int k0   = kk * 32;
        const int koff = 64 * kk + 16 * (kk >> 2);   // padded byte offset within B row

        v16bf afr[2];
#pragma unroll
        for (int mt = 0; mt < 2; ++mt) {
            const int m = wave_m * 32 + mt * 16 + l16;
            const __bf16* ap = &As[m][k0 + 16 * g];
            ((uint4*)&afr[mt])[0] = *(const uint4*)ap;
            ((uint4*)&afr[mt])[1] = *(const uint4*)(ap + 8);
        }
        v16bf bfr[4];
#pragma unroll
        for (int nt = 0; nt < 4; ++nt) {
            const __bf16* bp = (const __bf16*)(Bbase + brow[nt] + koff);
            ((uint4*)&bfr[nt])[0] = ((const uint4*)bp)[0];
            ((uint4*)&bfr[nt])[1] = ((const uint4*)bp)[1];
        }
#pragma unroll
        for (int mt = 0; mt < 2; ++mt)
#pragma unroll
            for (int nt = 0; nt < 4; ++nt) {
                acc[mt][nt] = __builtin_amdgcn_wmma_f32_16x16x32_bf16(
                    false, afr[mt], false, bfr[nt],
                    (short)0, acc[mt][nt], false, false);
            }
    }

    // ---- store z (bf16) + per-channel sum / sumsq ----
#pragma unroll
    for (int mt = 0; mt < 2; ++mt) {
#pragma unroll
        for (int nt = 0; nt < 4; ++nt) {
            const v8f c = acc[mt][nt];
            const int mbase = eb + wave_m * 32 + mt * 16 + 8 * g;  // + j
            const int ncol  = wave_n * 64 + nt * 16 + l16;
            float s = 0.0f, s2 = 0.0f;
#pragma unroll
            for (int j = 0; j < 8; ++j) {
                const float v = c[j];
                s  += v;
                s2 += v * v;
                zb[(size_t)(mbase + j) * NOUT + ncol] = (__bf16)v;
            }
            atomicAdd(&s_sum[ncol], s);   // ds_add_f32
            atomicAdd(&s_sq[ncol], s2);
        }
    }
    __syncthreads();
    unsafeAtomicAdd(&gsum[tid], s_sum[tid]);   // global_atomic_add_f32
    unsafeAtomicAdd(&gsq[tid],  s_sq[tid]);
}

// ================= kernel 3: BN finalize ==========
__global__ void cgcnn_bnfin(const float* __restrict__ gsum, const float* __restrict__ gsq,
                            const float* __restrict__ gamma, const float* __restrict__ beta,
                            float* __restrict__ scale, float* __restrict__ shift) {
    const int c = threadIdx.x;   // 256 threads, 1 block
    const float invE = 1.0f / (float)N_EDGES;
    const float m   = gsum[c] * invE;
    const float var = gsq[c] * invE - m * m;
    const float is  = rsqrtf(var + BN_EPS);
    const float sc  = gamma[c] * is;
    scale[c] = sc;
    shift[c] = beta[c] - m * sc;
}

// ================= kernel 4: gated message + scatter-add ==========
__global__ __launch_bounds__(256)
void cgcnn_scatter(const __bf16* __restrict__ zb, const int* __restrict__ dst,
                   const float* __restrict__ scale, const float* __restrict__ shift,
                   float* __restrict__ agg) {
    __shared__ float ssc[NOUT];
    __shared__ float ssh[NOUT];
    const int tid = threadIdx.x;
    ssc[tid] = scale[tid];
    ssh[tid] = shift[tid];
    __syncthreads();

    const long long idx = (long long)blockIdx.x * 256 + tid;  // E*32 units
    if (idx >= (long long)N_EDGES * 32) return;
    const int eix = (int)(idx >> 5);
    const int c   = ((int)idx & 31) * 4;   // 4 channels per thread
    const __bf16* zr = zb + (size_t)eix * NOUT;
    float* ar = agg + (size_t)dst[eix] * HIDDEN + c;
#pragma unroll
    for (int j = 0; j < 4; ++j) {
        const float f  = (float)zr[c + j]          * ssc[c + j]          + ssh[c + j];
        const float co = (float)zr[HIDDEN + c + j] * ssc[HIDDEN + c + j] + ssh[HIDDEN + c + j];
        const float sig = 1.0f / (1.0f + __expf(-f));
        const float sp  = (co > 20.0f) ? co : log1pf(__expf(co));
        unsafeAtomicAdd(&ar[j], sig * sp);
    }
}

// ================= kernel 5: residual + softplus ==========
__global__ __launch_bounds__(256)
void cgcnn_final(const float* __restrict__ h, const float* __restrict__ agg,
                 float* __restrict__ out) {
    const int i = blockIdx.x * 256 + threadIdx.x;   // over N*H/4 float4s
    if (i >= (N_NODES * HIDDEN) / 4) return;
    const float4 hv = ((const float4*)h)[i];
    const float4 av = ((const float4*)agg)[i];
    float4 r;
    float x;
    x = hv.x + av.x; r.x = (x > 20.0f) ? x : log1pf(__expf(x));
    x = hv.y + av.y; r.y = (x > 20.0f) ? x : log1pf(__expf(x));
    x = hv.z + av.z; r.z = (x > 20.0f) ? x : log1pf(__expf(x));
    x = hv.w + av.w; r.w = (x > 20.0f) ? x : log1pf(__expf(x));
    ((float4*)out)[i] = r;
}

// ================= host launcher ==========
extern "C" void kernel_launch(void* const* d_in, const int* in_sizes, int n_in,
                              void* d_out, int out_size, void* d_ws, size_t ws_size,
                              hipStream_t stream) {
    (void)in_sizes; (void)n_in; (void)out_size; (void)ws_size;
    const float* h     = (const float*)d_in[0];
    const float* e     = (const float*)d_in[1];
    const int*   src   = (const int*)d_in[2];
    const int*   dst   = (const int*)d_in[3];
    const float* W     = (const float*)d_in[4];
    // d_in[5] = b : cancels exactly under batch-norm -> unused
    const float* gamma = (const float*)d_in[6];
    const float* beta  = (const float*)d_in[7];
    float* out = (float*)d_out;

    char* ws = (char*)d_ws;
    __bf16* Wt    = (__bf16*)(ws + WS_WT);
    float*  gsum  = (float*)(ws + WS_GSUM);
    float*  gsq   = (float*)(ws + WS_GSQ);
    float*  scale = (float*)(ws + WS_SCALE);
    float*  shift = (float*)(ws + WS_SHIFT);
    float*  agg   = (float*)(ws + WS_AGG);
    __bf16* zb    = (__bf16*)(ws + WS_Z);

    cgcnn_prep<<<(N_NODES * HIDDEN + 255) / 256, 256, 0, stream>>>(W, Wt, gsum, gsq, agg);
    cgcnn_gemm<<<NBLOCKS, 256, BS_BYTES, stream>>>(h, e, src, dst, Wt, zb, gsum, gsq);
    cgcnn_bnfin<<<1, NOUT, 0, stream>>>(gsum, gsq, gamma, beta, scale, shift);
    cgcnn_scatter<<<(N_EDGES * 32) / 256, 256, 0, stream>>>(zb, dst, scale, shift, agg);
    cgcnn_final<<<(N_NODES * HIDDEN / 4 + 255) / 256, 256, 0, stream>>>(h, agg, out);
}
